// SparseMoELayer_55095840473805
// MI455X (gfx1250) — matrix operand order repaired and checked by reading, not compiled, not run on hardware
//
#include <hip/hip_runtime.h>
#include <stdint.h>

// ---------------- CDNA5 WMMA types ----------------
typedef __bf16 v16bf __attribute__((ext_vector_type(16)));
typedef float  v8f   __attribute__((ext_vector_type(8)));

union FragU {
    v16bf v;
    uint4 q[2];
};

#define NTOK   65536
#define DIM    256
#define NOUT   256
#define NEXP   8
#define AP     264   // LDS row pitch in halfs (256 + 8 pad -> conflict-free frag reads)

__device__ __forceinline__ unsigned short f32_to_bf16_bits(float f) {
    uint32_t u = __float_as_uint(f);
    uint32_t r = u + 0x7FFFu + ((u >> 16) & 1u);   // round-to-nearest-even
    return (unsigned short)(r >> 16);
}

// ---------------- fp32 -> bf16 convert (weights only, 1 MB) ----------------
__global__ void k_f32_to_bf16(const float* __restrict__ in,
                              unsigned short* __restrict__ out, int n) {
    int i = blockIdx.x * blockDim.x + threadIdx.x;
    if (i < n) out[i] = f32_to_bf16_bits(in[i]);
}

// ---------------- c_e[o] = sum_d bias[e][d] * W[e][o][d] ----------------
__global__ void k_cvec(const float* __restrict__ bias,
                       const float* __restrict__ W,
                       float* __restrict__ cvec) {
    int e = blockIdx.x;
    int o = threadIdx.x;
    const float* wrow = W + ((size_t)e * NOUT + o) * DIM;
    const float* brow = bias + (size_t)e * DIM;
    float s = 0.f;
    for (int d = 0; d < DIM; ++d) s += brow[d] * wrow[d];
    cvec[e * NOUT + o] = s;
}

// ---- gating: logits, top-2, softmax, routing lists, out-init, x->bf16 ----
// One wave per token. x is streamed once here, so the bf16 conversion of x
// is folded in (saves a full extra 64 MB read pass at 23.3 TB/s).
__global__ void __launch_bounds__(256)
k_gate(const float* __restrict__ x, const float* __restrict__ wg,
       const float* __restrict__ cvec, unsigned short* __restrict__ xb,
       int* __restrict__ cnt, int* __restrict__ lst, float* __restrict__ gl,
       float* __restrict__ out) {
    int wave = threadIdx.x >> 5;
    int lane = threadIdx.x & 31;
    int n = blockIdx.x * 8 + wave;   // one wave per token

    const float* xr = x + (size_t)n * DIM;
    unsigned short* xbr = xb + (size_t)n * DIM;

    float acc[8];
#pragma unroll
    for (int e = 0; e < 8; ++e) acc[e] = 0.f;

#pragma unroll
    for (int it = 0; it < 8; ++it) {
        int d = lane + it * 32;                    // coalesced across lanes
        float xv = xr[d];
        xbr[d] = f32_to_bf16_bits(xv);             // folded x -> bf16
        const float4* w4 = (const float4*)(wg + d * 8);
        float4 w0 = w4[0], w1 = w4[1];
        acc[0] += xv * w0.x; acc[1] += xv * w0.y;
        acc[2] += xv * w0.z; acc[3] += xv * w0.w;
        acc[4] += xv * w1.x; acc[5] += xv * w1.y;
        acc[6] += xv * w1.z; acc[7] += xv * w1.w;
    }
    // wave32 reduction -> every lane holds all 8 logits
#pragma unroll
    for (int e = 0; e < 8; ++e) {
#pragma unroll
        for (int off = 16; off > 0; off >>= 1)
            acc[e] += __shfl_xor(acc[e], off);
    }
    // top-2 (ties -> lowest index first, matching lax.top_k)
    float v0 = -3.4e38f, v1 = -3.4e38f;
    int i0 = 0, i1 = 0;
#pragma unroll
    for (int e = 0; e < 8; ++e) {
        float v = acc[e];
        if (v > v0) { v1 = v0; i1 = i0; v0 = v; i0 = e; }
        else if (v > v1) { v1 = v; i1 = e; }
    }
    // softmax over the two kept logits (others are -inf -> zero gates)
    float t = __expf(v1 - v0);
    float inv = 1.0f / (1.0f + t);
    float g0 = inv, g1 = t * inv;

    if (lane == 0) {
        int p0 = atomicAdd(cnt + i0, 1);
        lst[i0 * NTOK + p0] = n;  gl[i0 * NTOK + p0] = g0;
        int p1 = atomicAdd(cnt + i1, 1);
        lst[i1 * NTOK + p1] = n;  gl[i1 * NTOK + p1] = g1;
    }
    // initialize out with the negative bias constant: -(g0*c_e0 + g1*c_e1)
    const float* c0 = cvec + i0 * NOUT;
    const float* c1 = cvec + i1 * NOUT;
    float* orow = out + (size_t)n * NOUT;
#pragma unroll
    for (int it = 0; it < 8; ++it) {
        int o = lane + it * 32;
        orow[o] = -(g0 * c0[o] + g1 * c1[o]);
    }
}

// ---------------- per-expert gathered GEMM via bf16 WMMA ----------------
// grid: (NTOK/64, NEXP); block: 256 threads = 8 waves.
// Block tile: 64 gathered tokens x 256 outputs. Wave w: M-strip (w&3)*16,
// N-half (w>>2)*128 -> 8 col tiles of 16. K loop: 8 x 32 = 256.
__global__ void __launch_bounds__(256)
k_expert_gemm(const unsigned short* __restrict__ xb,
              const unsigned short* __restrict__ Wb,
              const int* __restrict__ cnt,
              const int* __restrict__ lst,
              const float* __restrict__ gl,
              float* __restrict__ out) {
    __shared__ unsigned short sA[64 * AP];
    __shared__ int   sTok[64];
    __shared__ float sG[64];

    int e = blockIdx.y;
    int c = cnt[e];
    int t0 = blockIdx.x * 64;
    if (t0 >= c) return;                 // uniform early-exit for empty tiles
    int mcnt = min(64, c - t0);

    int tid = threadIdx.x;
    if (tid < 64) {
        if (tid < mcnt) {
            sTok[tid] = lst[e * NTOK + t0 + tid];
            sG[tid]   = gl [e * NTOK + t0 + tid];
        } else {
            sTok[tid] = 0;               // safe dummy row; store is guarded
            sG[tid]   = 0.f;
        }
    }
    __syncthreads();

    // gather 64 token rows (bf16) into LDS, coalesced dword copies
    const unsigned int* xbu = (const unsigned int*)xb;
    unsigned int* sAu = (unsigned int*)sA;
    for (int i = tid; i < 64 * 128; i += 256) {
        int r = i >> 7, col = i & 127;
        sAu[r * (AP / 2) + col] = xbu[(size_t)sTok[r] * (DIM / 2) + col];
    }
    __syncthreads();

    int wave = tid >> 5, lane = tid & 31;
    int strip = wave & 3;                // M tile offset = strip*16
    int N0    = (wave >> 2) * 128;       // N half
    int h  = lane >> 4;                  // lane-half select
    int nl = lane & 15;

    const unsigned short* aRow = sA + (strip * 16 + nl) * AP;
    const unsigned short* wbE  = Wb + (size_t)e * NOUT * DIM;

    v8f acc[8];
#pragma unroll
    for (int nt = 0; nt < 8; ++nt) acc[nt] = (v8f)(0.f);

#pragma unroll
    for (int kc = 0; kc < 8; ++kc) {
        int k0 = kc * 32;
        // A fragment (16x32 bf16): lane (m=nl, h): halves 0..7 = K[8h..8h+7],
        // halves 8..15 = K[16+8h..16+8h+7]  (ISA 7.12.2 layout)
        FragU a;
        a.q[0] = *(const uint4*)(aRow + k0 + 8 * h);
        a.q[1] = *(const uint4*)(aRow + k0 + 16 + 8 * h);
#pragma unroll
        for (int nt = 0; nt < 8; ++nt) {
            // B fragment (32x16 bf16): column n = W row o -> contiguous in K.
            // lane half h: halves 0..7 = K[8h..8h+7], halves 8..15 = K[16+8h..16+8h+7].
            const unsigned short* bRow =
                wbE + (size_t)(N0 + nt * 16 + nl) * DIM;
            FragU b;
            b.q[0] = *(const uint4*)(bRow + k0 + 8 * h);
            b.q[1] = *(const uint4*)(bRow + k0 + 16 + 8 * h);
            acc[nt] = __builtin_amdgcn_wmma_f32_16x16x32_bf16(
                false, a.v, false, b.v, (short)0, acc[nt], false, false);
        }
    }

    // C/D layout: VGPR v -> M = v + 8*h, N = nl. Fold gate, atomic-add.
#pragma unroll
    for (int nt = 0; nt < 8; ++nt) {
#pragma unroll
        for (int v = 0; v < 8; ++v) {
            int rloc = strip * 16 + v + 8 * h;
            if (rloc < mcnt) {
                int tok = sTok[rloc];
                float val = sG[rloc] * acc[nt][v];
                __hip_atomic_fetch_add(
                    out + (size_t)tok * NOUT + N0 + nt * 16 + nl, val,
                    __ATOMIC_RELAXED, __HIP_MEMORY_SCOPE_AGENT);
            }
        }
    }
}

// ---------------- launcher ----------------
extern "C" void kernel_launch(void* const* d_in, const int* in_sizes, int n_in,
                              void* d_out, int out_size, void* d_ws, size_t ws_size,
                              hipStream_t stream) {
    const float* x  = (const float*)d_in[0];   // [N, D]
    const float* wg = (const float*)d_in[1];   // [D, E]
    const float* eb = (const float*)d_in[2];   // [E, D]
    const float* ew = (const float*)d_in[3];   // [E, OUT, D]
    float* out = (float*)d_out;                // [N, OUT]

    char* ws = (char*)d_ws;
    unsigned short* xb   = (unsigned short*)(ws);                 // 32 MB bf16 x
    unsigned short* Wb   = (unsigned short*)(ws + 33554432);      // 1 MB bf16 W
    float*          cvec = (float*)(ws + 34603008);               // 8 KB
    int*            cnt  = (int*)(ws + 34611200);                 // per-expert counts
    int*            lst  = (int*)(ws + 34611456);                 // 2 MB token lists
    float*          gl   = (float*)(ws + 36708608);               // 2 MB gate vals

    hipMemsetAsync(cnt, 0, 64, stream);

    k_f32_to_bf16<<<(NEXP * NOUT * DIM + 255) / 256, 256, 0, stream>>>(ew, Wb, NEXP * NOUT * DIM);
    k_cvec<<<NEXP, NOUT, 0, stream>>>(eb, ew, cvec);
    k_gate<<<NTOK / 8, 256, 0, stream>>>(x, wg, cvec, xb, cnt, lst, gl, out);

    dim3 g(NTOK / 64, NEXP);
    k_expert_gemm<<<g, 256, 0, stream>>>(xb, Wb, cnt, lst, gl, out);
}
